// InteractingLayer_53953379173118
// MI455X (gfx1250) — compile-verified, matrix operand order checked by reading, not run on hardware
//
#include <hip/hip_runtime.h>

typedef __attribute__((ext_vector_type(16))) _Float16 v16h;
typedef __attribute__((ext_vector_type(8)))  _Float16 v8h;
typedef __attribute__((ext_vector_type(8)))  float    v8f;

#define HEADS  4
#define EMB    64
#define ATT    32
#define OUTD   128
#define FIELDS 40
#define FPAD   48

// f16 copies of the four projection weight matrices, rebuilt every launch.
__device__ __align__(16) _Float16 g_Wh[4][OUTD][EMB];

__global__ void cvt_weights_kernel(const float* __restrict__ Wq,
                                   const float* __restrict__ Wk,
                                   const float* __restrict__ Wv,
                                   const float* __restrict__ Wr) {
  int idx = blockIdx.x * blockDim.x + threadIdx.x;  // 0 .. 4*128*64-1
  int p = idx >> 13;                                // 8192 elems per matrix
  int r = idx & 8191;
  const float* W = (p == 0) ? Wq : (p == 1) ? Wk : (p == 2) ? Wv : Wr;
  ((_Float16*)g_Wh)[idx] = (_Float16)W[r];
}

// Assemble one WMMA operand fragment for this lane: two contiguous 8-half
// chunks (K = base..+7 and base+16..+23), per the CDNA5 16-bit A/B VGPR
// layout. `p` must be 16-byte aligned -> two ds/global _b128 loads.
__device__ __forceinline__ v16h ldfrag(const _Float16* p) {
  v8h lo = *(const v8h*)(p);
  v8h hi = *(const v8h*)(p + 16);
  return __builtin_shufflevector(lo, hi, 0, 1, 2, 3, 4, 5, 6, 7,
                                          8, 9, 10, 11, 12, 13, 14, 15);
}

__device__ __forceinline__ v8f wmma_f16(v16h a, v16h b, v8f c) {
  return __builtin_amdgcn_wmma_f32_16x16x32_f16(false, a, false, b,
                                                (short)0, c, false, false);
}

__global__ void __launch_bounds__(256)
interacting_layer_kernel(const float* __restrict__ x,
                         const float* __restrict__ bq, const float* __restrict__ bk,
                         const float* __restrict__ bv, const float* __restrict__ br,
                         float* __restrict__ out) {
  // 64 KB static LDS, partitioned (units: halves):
  //   [0,12288)      P buffer [4][48][64]  (aliases xh [48][64] during stage A)
  //   [12288,18432)  Qh [4][48][32]
  //   [18432,24576)  Kh [4][48][32]
  //   [24576,32768)  Vt [4][32][64]  (V transposed: [head][d][g])
  __shared__ __align__(16) _Float16 smem[32768];
  _Float16* Pbuf = smem;
  _Float16* Qh   = smem + 12288;
  _Float16* Kh   = smem + 18432;
  _Float16* Vt   = smem + 24576;
  _Float16* xh   = Pbuf;  // alias: x (f16) lives here until stage B

  const int b    = blockIdx.x;
  const int tid  = threadIdx.x;
  const int w    = tid >> 5;        // wave id 0..7 (wave32)
  const int lane = tid & 31;
  const int l16  = lane & 15;
  const int half = lane >> 4;
  const int koff = half * 8;        // lane-half K offset within a 32-K window

  // ---- Load x row-block into LDS as f16 (rows 40..47 zeroed) ----
  const float* xb = x + (size_t)b * FIELDS * EMB;
  for (int i = tid; i < FPAD * EMB; i += 256) {
    int f = i >> 6, e = i & 63;
    float v = (f < FIELDS) ? xb[f * EMB + e] : 0.0f;
    xh[i] = (_Float16)v;
  }
  // Zero Vt pad columns g in [40,64) so the K=64 PV GEMM sees exact zeros.
  for (int i = tid; i < HEADS * ATT * 24; i += 256) {
    int hd = i / 24;
    int g  = 40 + i % 24;
    Vt[hd * 64 + g] = (_Float16)0.0f;
  }
  __syncthreads();

  // ---- Stage A: Q,K,V,R projections (96 tiles of 16x16, K=64) ----
  // Tile ownership j = w*3+i (m = j/8, n = j%8) is identical for all four
  // projections and for stage C, so residual tiles stay in this wave's VGPRs.
  // A-fragments (x rows) are loaded once and reused by all four projections;
  // p is compile-time so every store path is straight-line code.
  v8f racc[3];
#pragma unroll
  for (int i = 0; i < 3; ++i) {
    const int j = w * 3 + i, m = j >> 3, n = j & 7;
    const int h = n >> 1, d0 = (n & 1) * 16;
    const int fr0 = m * 16 + half * 8;                      // M = j8 + 8*half
    const _Float16* arow = xh + (m * 16 + l16) * EMB;
    const v16h a0 = ldfrag(arow + koff);
    const v16h a1 = ldfrag(arow + 32 + koff);

#pragma unroll
    for (int p = 0; p < 4; ++p) {
      const _Float16* brow = &g_Wh[p][n * 16 + l16][0];
      v8f c = {};
      c = wmma_f16(a0, ldfrag(brow + koff), c);
      c = wmma_f16(a1, ldfrag(brow + 32 + koff), c);
      const float* bias = (p == 0) ? bq : (p == 1) ? bk : (p == 2) ? bv : br;
      const float bvv = bias[n * 16 + l16];
      if (p == 0) {                       // Q -> [h][f][d]
#pragma unroll
        for (int j8 = 0; j8 < 8; ++j8)
          Qh[(h * FPAD + fr0 + j8) * ATT + d0 + l16] = (_Float16)(c[j8] + bvv);
      } else if (p == 1) {                // K -> [h][g][d]
#pragma unroll
        for (int j8 = 0; j8 < 8; ++j8)
          Kh[(h * FPAD + fr0 + j8) * ATT + d0 + l16] = (_Float16)(c[j8] + bvv);
      } else if (p == 2) {                // V -> transposed [h][d][g], packed b128
        v8h hv;
#pragma unroll
        for (int j8 = 0; j8 < 8; ++j8) hv[j8] = (_Float16)(c[j8] + bvv);
        *(v8h*)&Vt[(h * ATT + d0 + l16) * 64 + fr0] = hv;
      } else {                            // residual stays in registers
#pragma unroll
        for (int j8 = 0; j8 < 8; ++j8) c[j8] += bvv;
        racc[i] = c;
      }
    }
  }
  __syncthreads();

  // ---- Stage B: S = Q K^T * 1/sqrt(64)  (36 tiles, one K=32 WMMA each) ----
  for (int t = w; t < 36; t += 8) {
    int h = t / 9, rm = t % 9, mi = rm / 3, ni = rm % 3;
    const _Float16* arow = Qh + (h * FPAD + mi * 16 + l16) * ATT;
    const _Float16* brow = Kh + (h * FPAD + ni * 16 + l16) * ATT;
    v8f c = {};
    c = wmma_f16(ldfrag(arow + koff), ldfrag(brow + koff), c);
#pragma unroll
    for (int j8 = 0; j8 < 8; ++j8) {
      int fr = mi * 16 + j8 + half * 8;
      Pbuf[(h * FPAD + fr) * 64 + ni * 16 + l16] = (_Float16)(c[j8] * 0.125f);
    }
  }
  __syncthreads();

  // ---- Softmax over g (40 valid cols); zero pad cols 40..63 ----
  // Rows are 128B-aligned: vectorized v8h row I/O (5+5 b128 ops + 3 pad).
  if (tid < HEADS * FPAD) {
    int h = tid / FPAD, f = tid % FPAD;
    _Float16* row = Pbuf + (h * FPAD + f) * 64;
    v8h rr[5];
#pragma unroll
    for (int k = 0; k < 5; ++k) rr[k] = *(const v8h*)(row + k * 8);
    float rv[40];
    float mx = -1e30f;
#pragma unroll
    for (int g = 0; g < 40; ++g) { rv[g] = (float)rr[g >> 3][g & 7]; mx = fmaxf(mx, rv[g]); }
    float s = 0.0f;
#pragma unroll
    for (int g = 0; g < 40; ++g) { rv[g] = __expf(rv[g] - mx); s += rv[g]; }
    float inv = 1.0f / s;
#pragma unroll
    for (int k = 0; k < 5; ++k) {
      v8h ov;
#pragma unroll
      for (int e = 0; e < 8; ++e) ov[e] = (_Float16)(rv[k * 8 + e] * inv);
      *(v8h*)(row + k * 8) = ov;
    }
    const v8h zz = {};
#pragma unroll
    for (int k = 5; k < 8; ++k) *(v8h*)(row + k * 8) = zz;
  }
  __syncthreads();

  // ---- Stage C: O = P V, add register-resident residual, ReLU, store ----
  float* ob = out + (size_t)b * FIELDS * OUTD;
#pragma unroll
  for (int i = 0; i < 3; ++i) {
    const int j = w * 3 + i, m = j >> 3, n = j & 7;
    const int h = n >> 1, d0 = (n & 1) * 16;
    const _Float16* arow = Pbuf + (h * FPAD + m * 16 + l16) * 64;   // A: P rows (K=64)
    const _Float16* brow = Vt + (h * ATT + d0 + l16) * 64;          // B: V^T rows
    v8f c = {};
    c = wmma_f16(ldfrag(arow + koff),      ldfrag(brow + koff),      c);
    c = wmma_f16(ldfrag(arow + 32 + koff), ldfrag(brow + 32 + koff), c);
#pragma unroll
    for (int j8 = 0; j8 < 8; ++j8) {
      int fr = m * 16 + j8 + half * 8;
      if (fr < FIELDS) {
        float v = c[j8] + racc[i][j8];
        ob[fr * OUTD + n * 16 + l16] = fmaxf(v, 0.0f);
      }
    }
  }
}

extern "C" void kernel_launch(void* const* d_in, const int* in_sizes, int n_in,
                              void* d_out, int out_size, void* d_ws, size_t ws_size,
                              hipStream_t stream) {
  const float* x  = (const float*)d_in[0];
  const float* Wq = (const float*)d_in[1];
  const float* bq = (const float*)d_in[2];
  const float* Wk = (const float*)d_in[3];
  const float* bk = (const float*)d_in[4];
  const float* Wv = (const float*)d_in[5];
  const float* bv = (const float*)d_in[6];
  const float* Wr = (const float*)d_in[7];
  const float* br = (const float*)d_in[8];
  float* out = (float*)d_out;
  (void)d_ws; (void)ws_size; (void)n_in; (void)out_size;

  // 4*128*64 = 32768 weight elements -> f16 device-global, built every launch.
  cvt_weights_kernel<<<128, 256, 0, stream>>>(Wq, Wk, Wv, Wr);

  int B = in_sizes[0] / (FIELDS * EMB);   // 16384
  interacting_layer_kernel<<<B, 256, 0, stream>>>(x, bq, bk, bv, br, out);
}
